// EncoderBlock_4063039062794
// MI455X (gfx1250) — matrix operand order repaired and checked by reading, not compiled, not run on hardware
//
#include <hip/hip_runtime.h>

typedef __bf16 bf16_t;
typedef __attribute__((ext_vector_type(16))) __bf16 bf16x16;
typedef __attribute__((ext_vector_type(8)))  __bf16 bf16x8;
typedef __attribute__((ext_vector_type(8)))  float   f32x8;

#define NN_C 256
#define NN_P 64
#define NN_K 32

// ---------------------------------------------------------------------------
// WMMA fragment layouts (CDNA5 ISA 7.12.2, 16-bit operands, wave32):
//   A (16xK): lane m = lane&15, half = lane>>4; elems e<8 -> k = 8*half+e,
//             e>=8 -> k = 16+8*half+(e-8)  => two contiguous 8-elem runs.
//   B (Kx16): lane n = lane&15, half = lane>>4; same k order as A:
//             e<8 -> k = 8*half+e, e>=8 -> k = 16+8*half+(e-8).
// A lives in LDS (bf16 row-major) -> 2x ds_load_b128 per fragment.
// B lives in global as bf16 TRANSPOSED Bt[n][k] -> 2x global_load_b128.
// ---------------------------------------------------------------------------
__device__ __forceinline__ f32x8 wmma_lt(const bf16_t* A, int lda,
                                         const bf16_t* Bt, int Kdim, int n0,
                                         f32x8 acc) {
  const int lane = threadIdx.x & 31;
  const int half = lane >> 4;
  const int mn = lane & 15;
  const bf16_t* arow = A + mn * lda + 8 * half;
  const bf16_t* brow = Bt + (size_t)(n0 + mn) * Kdim + 8 * half;
  __builtin_prefetch((const void*)brow, 0, 3);  // global_prefetch_b8
  for (int k0 = 0; k0 < Kdim; k0 += 32) {
    bf16x8 a0 = *(const bf16x8*)(arow + k0);
    bf16x8 a1 = *(const bf16x8*)(arow + k0 + 16);
    bf16x8 b0 = *(const bf16x8*)(brow + k0);
    bf16x8 b1 = *(const bf16x8*)(brow + k0 + 16);
    bf16x16 a = __builtin_shufflevector(a0, a1, 0, 1, 2, 3, 4, 5, 6, 7, 8, 9, 10, 11, 12, 13, 14, 15);
    bf16x16 b = __builtin_shufflevector(b0, b1, 0, 1, 2, 3, 4, 5, 6, 7, 8, 9, 10, 11, 12, 13, 14, 15);
    acc = __builtin_amdgcn_wmma_f32_16x16x32_bf16(false, a, false, b, (short)0, acc, false, false);
  }
  return acc;
}

__device__ __forceinline__ void tile_store_f32(float* D, int ldd, f32x8 acc) {
  const int lane = threadIdx.x & 31;
  const int half = lane >> 4;
  const int n = lane & 15;
#pragma unroll
  for (int r = 0; r < 8; ++r) D[(r + 8 * half) * ldd + n] = acc[r];
}
__device__ __forceinline__ void tile_store_bf16(bf16_t* D, int ldd, f32x8 acc) {
  const int lane = threadIdx.x & 31;
  const int half = lane >> 4;
  const int n = lane & 15;
#pragma unroll
  for (int r = 0; r < 8; ++r) D[(r + 8 * half) * ldd + n] = (bf16_t)acc[r];
}

__device__ __forceinline__ float gelu_f(float x) {
  const float c = 0.7978845608028654f;  // sqrt(2/pi)
  return 0.5f * x * (1.f + tanhf(c * (x + 0.044715f * x * x * x)));
}

// ---------- Kernel 0: pack a weight matrix W[K][Nc] (f32) -> Bt[Nc][K] (bf16) ----------
__global__ void __launch_bounds__(256) pack_wT_kernel(const float* __restrict__ W,
                                                      bf16_t* __restrict__ out,
                                                      int K, int Nc) {
  int i = blockIdx.x * 256 + threadIdx.x;
  if (i >= K * Nc) return;
  int n = i / K, k = i - n * K;
  out[i] = (bf16_t)W[(size_t)k * Nc + n];
}

// ---------- Kernel 1: frames ----------
__global__ void __launch_bounds__(256) frames_kernel(const float* __restrict__ pos,
                                                     float* __restrict__ Rg,
                                                     float* __restrict__ cag, int N) {
  int n = blockIdx.x * blockDim.x + threadIdx.x;
  if (n >= N) return;
  const float* p = pos + (size_t)n * 15;
  float ax = p[3], ay = p[4], az = p[5];
  float e1x = p[6] - ax, e1y = p[7] - ay, e1z = p[8] - az;
  float inv = 1.f / (sqrtf(e1x * e1x + e1y * e1y + e1z * e1z) + 1e-6f);
  e1x *= inv; e1y *= inv; e1z *= inv;
  float ux = p[0] - ax, uy = p[1] - ay, uz = p[2] - az;
  float dt = ux * e1x + uy * e1y + uz * e1z;
  float e2x = ux - dt * e1x, e2y = uy - dt * e1y, e2z = uz - dt * e1z;
  inv = 1.f / (sqrtf(e2x * e2x + e2y * e2y + e2z * e2z) + 1e-6f);
  e2x *= inv; e2y *= inv; e2z *= inv;
  float e3x = e1y * e2z - e1z * e2y;
  float e3y = e1z * e2x - e1x * e2z;
  float e3z = e1x * e2y - e1y * e2x;
  float* r = Rg + (size_t)n * 9;  // R[j][m] = e_m[j]
  r[0] = e1x; r[1] = e2x; r[2] = e3x;
  r[3] = e1y; r[4] = e2y; r[5] = e3y;
  r[6] = e1z; r[7] = e2z; r[8] = e3z;
  cag[n * 3 + 0] = ax; cag[n * 3 + 1] = ay; cag[n * 3 + 2] = az;
}

// ---------- Kernel 2: QKV GEMMs ----------
__global__ void __launch_bounds__(256) qkv_kernel(const float* __restrict__ x,
                                                  const bf16_t* __restrict__ wqT,
                                                  const bf16_t* __restrict__ wkT,
                                                  const bf16_t* __restrict__ wvT,
                                                  float* __restrict__ q, float* __restrict__ kk,
                                                  float* __restrict__ vv, int N) {
  __shared__ bf16_t xs[64][256];
  int row0 = blockIdx.x * 64;
  for (int i = threadIdx.x; i < 64 * 256; i += 256) {
    int r = i >> 8, c = i & 255;
    xs[r][c] = (bf16_t)x[(size_t)(row0 + r) * 256 + c];
  }
  __syncthreads();
  int wave = threadIdx.x >> 5;
  for (int t = wave; t < 192; t += 8) {
    int mat = t >> 6, rem = t & 63, rt = rem >> 4, ct = rem & 15;
    const bf16_t* W = (mat == 0) ? wqT : ((mat == 1) ? wkT : wvT);
    float* O = (mat == 0) ? q : ((mat == 1) ? kk : vv);
    f32x8 acc = {};
    acc = wmma_lt(&xs[rt * 16][0], 256, W, 256, ct * 16, acc);
    tile_store_f32(O + (size_t)(row0 + rt * 16) * 256 + ct * 16, 256, acc);
  }
}

// ---------- Kernel 3: fused pair features + pair MLP + bias/pv + attention ----------
__global__ void __launch_bounds__(128) pair_attn_kernel(
    const float* __restrict__ pos, const float* __restrict__ maskp,
    const int* __restrict__ neighbours, const int* __restrict__ resi,
    const int* __restrict__ chain, const int* __restrict__ batch,
    const float* __restrict__ Rg, const float* __restrict__ cag,
    const float* __restrict__ w_relpos, const float* __restrict__ w_dist,
    const float* __restrict__ w_dir, const float* __restrict__ w_rot,
    const float* __restrict__ w_pvec, const float* __restrict__ pln_s,
    const float* __restrict__ pln_b, const bf16_t* __restrict__ w_p1T,
    const float* __restrict__ b_p1, const bf16_t* __restrict__ w_p2T,
    const float* __restrict__ b_p2, const bf16_t* __restrict__ w_biasT,
    const bf16_t* __restrict__ w_pvT, const float* __restrict__ q,
    const float* __restrict__ kbuf, const float* __restrict__ vbuf,
    float* __restrict__ attn_out, int N) {
  // Aliased LDS regions (peak 60KB):
  // [0,16K)   pairF f32[64][64]   -> later h1 bf16[64][128]
  // [16K,24K) pairLN bf16[64][64] -> later pair2 bf16[64][64]
  // [24K,56K) pv bf16[64][256]
  // [56K,58K) biasL f32[64][8]
  __shared__ char smem[60 * 1024];
  float*  pairF  = (float*)smem;
  bf16_t* h1     = (bf16_t*)smem;
  bf16_t* pairLN = (bf16_t*)(smem + 16 * 1024);
  bf16_t* pair2  = (bf16_t*)(smem + 16 * 1024);
  bf16_t* pv     = (bf16_t*)(smem + 24 * 1024);
  float*  biasL  = (float*)(smem + 56 * 1024);

  const int tid = threadIdx.x;
  const int node0 = blockIdx.x * 2;

  // -------- Phase A: raw pair features (2 threads per edge, 32 feats each) --------
  {
    int edge = tid >> 1;
    int phalf = (tid & 1) * 32;
    int nd = edge >> 5, kk = edge & 31;
    int n = node0 + nd;
    int nbraw = neighbours[n * NN_K + kk];
    int nb = nbraw < 0 ? 0 : nbraw;
    bool same = (chain[n] == chain[nb]) && (batch[n] == batch[nb]);
    int rel = resi[nb] - resi[n];
    rel = rel < -32 ? -32 : (rel > 32 ? 32 : rel);
    rel += 32;
    if (!same) rel = 65;
    float cx = cag[n * 3], cy = cag[n * 3 + 1], cz = cag[n * 3 + 2];
    float dx = cag[nb * 3] - cx, dy = cag[nb * 3 + 1] - cy, dz = cag[nb * 3 + 2] - cz;
    float d = sqrtf(dx * dx + dy * dy + dz * dz);
    float rbf[16];
#pragma unroll
    for (int b = 0; b < 16; ++b) {
      float t = (d - 22.f * (float)b / 15.f) / 1.375f;
      rbf[b] = expf(-t * t);
    }
    const float* Rn = Rg + (size_t)n * 9;
    float loc[15], dir[15];
#pragma unroll
    for (int a = 0; a < 5; ++a) {
      float vx = pos[(size_t)nb * 15 + a * 3 + 0] - cx;
      float vy = pos[(size_t)nb * 15 + a * 3 + 1] - cy;
      float vz = pos[(size_t)nb * 15 + a * 3 + 2] - cz;
      float l0 = vx * Rn[0] + vy * Rn[3] + vz * Rn[6];
      float l1 = vx * Rn[1] + vy * Rn[4] + vz * Rn[7];
      float l2 = vx * Rn[2] + vy * Rn[5] + vz * Rn[8];
      loc[a * 3 + 0] = l0; loc[a * 3 + 1] = l1; loc[a * 3 + 2] = l2;
      float in2 = 1.f / (sqrtf(l0 * l0 + l1 * l1 + l2 * l2) + 1e-6f);
      dir[a * 3 + 0] = l0 * in2; dir[a * 3 + 1] = l1 * in2; dir[a * 3 + 2] = l2 * in2;
    }
    const float* Rb = Rg + (size_t)nb * 9;
    float rot[9];
#pragma unroll
    for (int i = 0; i < 3; ++i)
#pragma unroll
      for (int m = 0; m < 3; ++m)
        rot[i * 3 + m] = Rn[0 + i] * Rb[0 + m] + Rn[3 + i] * Rb[3 + m] + Rn[6 + i] * Rb[6 + m];
    for (int pp = 0; pp < 32; ++pp) {
      int p = phalf + pp;
      float v = w_relpos[rel * NN_P + p];
#pragma unroll
      for (int b = 0; b < 16; ++b) v += rbf[b] * w_dist[b * NN_P + p];
#pragma unroll
      for (int j = 0; j < 15; ++j) v += dir[j] * w_dir[j * NN_P + p];
#pragma unroll
      for (int j = 0; j < 9; ++j) v += rot[j] * w_rot[j * NN_P + p];
#pragma unroll
      for (int j = 0; j < 15; ++j) v += loc[j] * w_pvec[j * NN_P + p];
      pairF[edge * NN_P + p] = v;
    }
  }
  __syncthreads();

  // -------- LayerNorm over P=64 --------
  if (tid < 64) {
    float row[64], m = 0.f;
#pragma unroll
    for (int p = 0; p < 64; ++p) { row[p] = pairF[tid * 64 + p]; m += row[p]; }
    m *= (1.f / 64.f);
    float s = 0.f;
#pragma unroll
    for (int p = 0; p < 64; ++p) { float t = row[p] - m; s += t * t; }
    s = rsqrtf(s * (1.f / 64.f) + 1e-5f);
#pragma unroll
    for (int p = 0; p < 64; ++p)
      pairLN[tid * 64 + p] = (bf16_t)((row[p] - m) * s * pln_s[p] + pln_b[p]);
  }
  __syncthreads();

  const int wave = tid >> 5;
  const int lane = tid & 31;
  const int half = lane >> 4;
  const int mn = lane & 15;

  // -------- h1 = gelu(pairLN @ w_p1 + b_p1)  [64x64]@[64x128] --------
  for (int t = wave; t < 32; t += 4) {
    int rt = t >> 3, ct = t & 7;
    f32x8 acc = {};
    acc = wmma_lt(pairLN + rt * 16 * 64, 64, w_p1T, 64, ct * 16, acc);
#pragma unroll
    for (int r = 0; r < 8; ++r)
      h1[(rt * 16 + r + 8 * half) * 128 + ct * 16 + mn] =
          (bf16_t)gelu_f(acc[r] + b_p1[ct * 16 + mn]);
  }
  __syncthreads();

  // -------- pair2 = h1 @ w_p2 + b_p2  [64x128]@[128x64] --------
  for (int t = wave; t < 16; t += 4) {
    int rt = t >> 2, ct = t & 3;
    f32x8 acc = {};
    acc = wmma_lt(h1 + rt * 16 * 128, 128, w_p2T, 128, ct * 16, acc);
#pragma unroll
    for (int r = 0; r < 8; ++r)
      pair2[(rt * 16 + r + 8 * half) * 64 + ct * 16 + mn] =
          (bf16_t)(acc[r] + b_p2[ct * 16 + mn]);
  }
  __syncthreads();

  // -------- biasL = pair2 @ w_bias  [64x64]@[64x8] (N-guarded WMMA) --------
  if (wave < 4) {
    int rt = wave;
    const bf16_t* arow = pair2 + (rt * 16 + mn) * 64 + 8 * half;
    const bf16_t* brow = w_biasT + (size_t)(mn < 8 ? mn : 7) * 64 + 8 * half;
    f32x8 acc = {};
    for (int k0 = 0; k0 < 64; k0 += 32) {
      bf16x8 a0 = *(const bf16x8*)(arow + k0);
      bf16x8 a1 = *(const bf16x8*)(arow + k0 + 16);
      bf16x8 b0 = *(const bf16x8*)(brow + k0);
      bf16x8 b1 = *(const bf16x8*)(brow + k0 + 16);
      bf16x16 a = __builtin_shufflevector(a0, a1, 0, 1, 2, 3, 4, 5, 6, 7, 8, 9, 10, 11, 12, 13, 14, 15);
      bf16x16 b = __builtin_shufflevector(b0, b1, 0, 1, 2, 3, 4, 5, 6, 7, 8, 9, 10, 11, 12, 13, 14, 15);
      acc = __builtin_amdgcn_wmma_f32_16x16x32_bf16(false, a, false, b, (short)0, acc, false, false);
    }
    if (mn < 8) {
#pragma unroll
      for (int r = 0; r < 8; ++r) biasL[(rt * 16 + r + 8 * half) * 8 + mn] = acc[r];
    }
  }
  // -------- pv = pair2 @ w_pv  [64x64]@[64x256] --------
  for (int t = wave; t < 64; t += 4) {
    int rt = t >> 4, ct = t & 15;
    f32x8 acc = {};
    acc = wmma_lt(pair2 + rt * 16 * 64, 64, w_pvT, 64, ct * 16, acc);
    tile_store_bf16(pv + rt * 16 * 256 + ct * 16, 256, acc);
  }
  __syncthreads();

  // -------- fused neighbor attention (wave: node = wave&1, heads 4*(wave>>1)..) --------
  {
    int nd = wave & 1;
    int h0 = (wave >> 1) * 4;
    int n = node0 + nd;
    int nbraw = neighbours[n * NN_K + lane];
    int nb = nbraw < 0 ? 0 : nbraw;
    float pmask = maskp[n] * maskp[nb] * ((nbraw != -1) ? 1.f : 0.f);
    int edge = nd * 32 + lane;
    const float scale = 0.17677669529663687f;  // 1/sqrt(32)
    float att[4];
#pragma unroll
    for (int h = 0; h < 4; ++h) {
      const float* qp = q + (size_t)n * 256 + (h0 + h) * 32;
      const float* kp = kbuf + (size_t)nb * 256 + (h0 + h) * 32;
      float dot = 0.f;
#pragma unroll
      for (int dd = 0; dd < 32; ++dd) dot += qp[dd] * kp[dd];
      float lg = dot * scale + biasL[edge * 8 + h0 + h];
      att[h] = (pmask > 0.f) ? lg : -1e9f;
    }
#pragma unroll
    for (int h = 0; h < 4; ++h) {
      float m = att[h];
      for (int off = 16; off > 0; off >>= 1) m = fmaxf(m, __shfl_xor(m, off, 32));
      float e = expf(att[h] - m);
      float s = e;
      for (int off = 16; off > 0; off >>= 1) s += __shfl_xor(s, off, 32);
      att[h] = e / s;
    }
    // lanes switch role: lane = d
#pragma unroll
    for (int h = 0; h < 4; ++h) {
      float o = 0.f;
      for (int k2 = 0; k2 < 32; ++k2) {
        float a = __shfl(att[h], k2, 32);
        int nbk = __shfl(nb, k2, 32);
        int ed = nd * 32 + k2;
        o += a * (vbuf[(size_t)nbk * 256 + (h0 + h) * 32 + lane] +
                  (float)pv[ed * 256 + (h0 + h) * 32 + lane]);
      }
      attn_out[(size_t)n * 256 + (h0 + h) * 32 + lane] = o;
    }
  }
}

// ---------- Kernel 4: out projection + residual + LN1 ----------
__global__ void __launch_bounds__(256) outproj_kernel(
    const float* __restrict__ attn_out, const bf16_t* __restrict__ woT,
    const float* __restrict__ bo, const float* __restrict__ x,
    const float* __restrict__ finc, const float* __restrict__ maskp,
    const float* __restrict__ ln1_s, const float* __restrict__ ln1_b,
    float* __restrict__ out2, float* __restrict__ x1ln, int N) {
  __shared__ bf16_t As[32][256];
  __shared__ float upd[32][256];
  int row0 = blockIdx.x * 32;
  for (int i = threadIdx.x; i < 32 * 256; i += 256) {
    int r = i >> 8, c = i & 255;
    As[r][c] = (bf16_t)attn_out[(size_t)(row0 + r) * 256 + c];
  }
  __syncthreads();
  const int wave = threadIdx.x >> 5;
  const int lane = threadIdx.x & 31;
  const int half = lane >> 4;
  const int nn = lane & 15;
  for (int t = wave; t < 32; t += 8) {
    int rt = t >> 4, ct = t & 15;
    f32x8 acc = {};
    acc = wmma_lt(&As[rt * 16][0], 256, woT, 256, ct * 16, acc);
#pragma unroll
    for (int r = 0; r < 8; ++r) upd[rt * 16 + r + 8 * half][ct * 16 + nn] = acc[r];
  }
  __syncthreads();
  for (int rr = wave; rr < 32; rr += 8) {
    int n = row0 + rr;
    float mk = maskp[n];
    float vals[8], sum = 0.f;
#pragma unroll
    for (int j = 0; j < 8; ++j) {
      int c = lane + 32 * j;
      float u = (upd[rr][c] + bo[c]) * mk;
      out2[(size_t)n * 256 + c] = finc[(size_t)n * 256 + c] + u;
      vals[j] = x[(size_t)n * 256 + c] + u;
      sum += vals[j];
    }
    for (int off = 16; off > 0; off >>= 1) sum += __shfl_xor(sum, off, 32);
    float mean = sum / 256.f, var = 0.f;
#pragma unroll
    for (int j = 0; j < 8; ++j) { float t = vals[j] - mean; var += t * t; }
    for (int off = 16; off > 0; off >>= 1) var += __shfl_xor(var, off, 32);
    float inv = rsqrtf(var / 256.f + 1e-5f);
#pragma unroll
    for (int j = 0; j < 8; ++j) {
      int c = lane + 32 * j;
      x1ln[(size_t)n * 256 + c] = (vals[j] - mean) * inv * ln1_s[c] + ln1_b[c];
    }
  }
}

// ---------- Kernel 5: transition MLP (fused), accumulate features_inc, stage pre-LN2 ----------
__global__ void __launch_bounds__(256) transition_kernel(
    const float* __restrict__ x1ln, const float* __restrict__ pos,
    const float* __restrict__ Rg, const float* __restrict__ cag,
    const float* __restrict__ wu1, const float* __restrict__ bu1,
    const bf16_t* __restrict__ wu2T, const float* __restrict__ bu2,
    const bf16_t* __restrict__ w_updT, const bf16_t* __restrict__ w_gateT,
    const bf16_t* __restrict__ w_outT, const float* __restrict__ b_out,
    float* __restrict__ out2, float* __restrict__ ybuf, int N) {
  // [0,32K) g1/gu bf16[32][512]; [32K,48K) h bf16[32][256]; [48K,50K) lps f32[32][16]
  __shared__ char smem[50 * 1024];
  bf16_t* g1 = (bf16_t*)smem;
  bf16_t* hs = (bf16_t*)(smem + 32 * 1024);
  float* lps = (float*)(smem + 48 * 1024);
  const int tid = threadIdx.x;
  int row0 = blockIdx.x * 32;

  if (tid < 32) {
    int n = row0 + tid;
    float cx = cag[n * 3], cy = cag[n * 3 + 1], cz = cag[n * 3 + 2];
    const float* Rn = Rg + (size_t)n * 9;
#pragma unroll
    for (int a = 0; a < 5; ++a) {
      float vx = pos[(size_t)n * 15 + a * 3 + 0] - cx;
      float vy = pos[(size_t)n * 15 + a * 3 + 1] - cy;
      float vz = pos[(size_t)n * 15 + a * 3 + 2] - cz;
      lps[tid * 16 + a * 3 + 0] = vx * Rn[0] + vy * Rn[3] + vz * Rn[6];
      lps[tid * 16 + a * 3 + 1] = vx * Rn[1] + vy * Rn[4] + vz * Rn[7];
      lps[tid * 16 + a * 3 + 2] = vx * Rn[2] + vy * Rn[5] + vz * Rn[8];
    }
  }
  __syncthreads();
  // g1 = gelu(local_pos @ wu1 + bu1)  (K=15, VALU)
  for (int i = tid; i < 32 * 512; i += 256) {
    int r = i >> 9, c = i & 511;
    float s = bu1[c];
#pragma unroll
    for (int j = 0; j < 15; ++j) s += lps[r * 16 + j] * wu1[j * 512 + c];
    g1[r * 512 + c] = (bf16_t)gelu_f(s);
  }
  __syncthreads();

  const int wave = tid >> 5;
  const int lane = tid & 31;
  const int half = lane >> 4;
  const int nn = lane & 15;

  // h = x1ln + g1 @ wu2 + bu2  [32x512]@[512x256]
  for (int t = wave; t < 32; t += 8) {
    int rt = t >> 4, ct = t & 15;
    f32x8 acc = {};
    acc = wmma_lt(g1 + rt * 16 * 512, 512, wu2T, 512, ct * 16, acc);
#pragma unroll
    for (int r = 0; r < 8; ++r) {
      int rr = rt * 16 + r + 8 * half;
      int c = ct * 16 + nn;
      hs[rr * 256 + c] = (bf16_t)(x1ln[(size_t)(row0 + rr) * 256 + c] + acc[r] + bu2[c]);
    }
  }
  __syncthreads();

  // gu = gelu(h @ w_gate) * (h @ w_upd)  [32x256]@[256x512] (x2)
  for (int t = wave; t < 64; t += 8) {
    int rt = t >> 5, ct = t & 31;
    f32x8 au = {}, ag = {};
    au = wmma_lt(hs + rt * 16 * 256, 256, w_updT, 256, ct * 16, au);
    ag = wmma_lt(hs + rt * 16 * 256, 256, w_gateT, 256, ct * 16, ag);
#pragma unroll
    for (int r = 0; r < 8; ++r)
      g1[(rt * 16 + r + 8 * half) * 512 + ct * 16 + nn] = (bf16_t)(gelu_f(ag[r]) * au[r]);
  }
  __syncthreads();

  // update2 = gu @ w_out + b_out; out2 += update2; ybuf = x1ln + update2
  for (int t = wave; t < 32; t += 8) {
    int rt = t >> 4, ct = t & 15;
    f32x8 acc = {};
    acc = wmma_lt(g1 + rt * 16 * 512, 512, w_outT, 512, ct * 16, acc);
#pragma unroll
    for (int r = 0; r < 8; ++r) {
      int n = row0 + rt * 16 + r + 8 * half;
      int c = ct * 16 + nn;
      float u = acc[r] + b_out[c];
      out2[(size_t)n * 256 + c] += u;
      ybuf[(size_t)n * 256 + c] = x1ln[(size_t)n * 256 + c] + u;
    }
  }
}

// ---------- Kernel 6: final LN2 (wave per row) ----------
__global__ void __launch_bounds__(256) ln2_kernel(const float* __restrict__ ybuf,
                                                  const float* __restrict__ s,
                                                  const float* __restrict__ b,
                                                  float* __restrict__ out1, int N) {
  const int wave = threadIdx.x >> 5;
  const int lane = threadIdx.x & 31;
  int n = blockIdx.x * 8 + wave;
  if (n >= N) return;
  float vals[8], sum = 0.f;
#pragma unroll
  for (int j = 0; j < 8; ++j) { vals[j] = ybuf[(size_t)n * 256 + lane + 32 * j]; sum += vals[j]; }
  for (int off = 16; off > 0; off >>= 1) sum += __shfl_xor(sum, off, 32);
  float mean = sum / 256.f, var = 0.f;
#pragma unroll
  for (int j = 0; j < 8; ++j) { float t = vals[j] - mean; var += t * t; }
  for (int off = 16; off > 0; off >>= 1) var += __shfl_xor(var, off, 32);
  float inv = rsqrtf(var / 256.f + 1e-5f);
#pragma unroll
  for (int j = 0; j < 8; ++j) {
    int c = lane + 32 * j;
    out1[(size_t)n * 256 + c] = (vals[j] - mean) * inv * s[c] + b[c];
  }
}

extern "C" void kernel_launch(void* const* d_in, const int* in_sizes, int n_in,
                              void* d_out, int out_size, void* d_ws, size_t ws_size,
                              hipStream_t stream) {
  (void)n_in; (void)out_size; (void)ws_size;
  const float* x     = (const float*)d_in[0];
  const float* finc  = (const float*)d_in[1];
  const float* pos   = (const float*)d_in[2];
  const float* maskp = (const float*)d_in[3];
  const int*   nbrs  = (const int*)d_in[4];
  const int*   resi  = (const int*)d_in[5];
  const int*   chain = (const int*)d_in[6];
  const int*   batch = (const int*)d_in[7];
  const float* w_relpos = (const float*)d_in[8];
  const float* w_dist   = (const float*)d_in[9];
  const float* w_dir    = (const float*)d_in[10];
  const float* w_rot    = (const float*)d_in[11];
  const float* w_pvec   = (const float*)d_in[12];
  const float* pln_s    = (const float*)d_in[13];
  const float* pln_b    = (const float*)d_in[14];
  const float* w_p1     = (const float*)d_in[15];
  const float* b_p1     = (const float*)d_in[16];
  const float* w_p2     = (const float*)d_in[17];
  const float* b_p2     = (const float*)d_in[18];
  const float* wq       = (const float*)d_in[19];
  const float* wk       = (const float*)d_in[20];
  const float* wv       = (const float*)d_in[21];
  const float* w_bias   = (const float*)d_in[22];
  const float* w_pv     = (const float*)d_in[23];
  const float* wo       = (const float*)d_in[24];
  const float* bo       = (const float*)d_in[25];
  const float* ln1_s    = (const float*)d_in[26];
  const float* ln1_b    = (const float*)d_in[27];
  const float* wu1      = (const float*)d_in[28];
  const float* bu1      = (const float*)d_in[29];
  const float* wu2      = (const float*)d_in[30];
  const float* bu2      = (const float*)d_in[31];
  const float* w_upd    = (const float*)d_in[32];
  const float* w_gate   = (const float*)d_in[33];
  const float* w_out    = (const float*)d_in[34];
  const float* b_out    = (const float*)d_in[35];
  const float* ln2_s    = (const float*)d_in[36];
  const float* ln2_b    = (const float*)d_in[37];

  const int N = in_sizes[3];  // mask has N elements

  float* out1 = (float*)d_out;              // final x
  float* out2 = out1 + (size_t)N * 256;     // final features_inc

  float* ws   = (float*)d_ws;
  float* Rg   = ws;
  float* cag  = Rg + (size_t)N * 9;
  float* q    = cag + (size_t)N * 3;
  float* kb   = q + (size_t)N * 256;
  float* vb   = kb + (size_t)N * 256;
  float* ao   = vb + (size_t)N * 256;   // attn_out, later reused as ybuf
  float* x1ln = ao + (size_t)N * 256;

  // bf16 transposed-weight arena (packed once per launch; L2-resident)
  bf16_t* wbf     = (bf16_t*)(x1ln + (size_t)N * 256);
  bf16_t* wqT     = wbf;                  // [256][256]
  bf16_t* wkT     = wqT + 256 * 256;
  bf16_t* wvT     = wkT + 256 * 256;
  bf16_t* woT     = wvT + 256 * 256;
  bf16_t* w_p1T   = woT + 256 * 256;      // [128][64]
  bf16_t* w_p2T   = w_p1T + 64 * 128;     // [64][128]
  bf16_t* w_biasT = w_p2T + 128 * 64;     // [8][64]
  bf16_t* w_pvT   = w_biasT + 64 * 8;     // [256][64]
  bf16_t* wu2T    = w_pvT + 64 * 256;     // [256][512]
  bf16_t* w_updT  = wu2T + 512 * 256;     // [512][256]
  bf16_t* w_gateT = w_updT + 256 * 512;   // [512][256]
  bf16_t* w_outT  = w_gateT + 256 * 512;  // [256][512]

  auto packT = [&](const float* W, bf16_t* o, int K, int Nc) {
    int tot = K * Nc;
    pack_wT_kernel<<<(tot + 255) / 256, 256, 0, stream>>>(W, o, K, Nc);
  };
  packT(wq, wqT, 256, 256);
  packT(wk, wkT, 256, 256);
  packT(wv, wvT, 256, 256);
  packT(wo, woT, 256, 256);
  packT(w_p1, w_p1T, 64, 128);
  packT(w_p2, w_p2T, 128, 64);
  packT(w_bias, w_biasT, 64, 8);
  packT(w_pv, w_pvT, 64, 256);
  packT(wu2, wu2T, 512, 256);
  packT(w_upd, w_updT, 256, 512);
  packT(w_gate, w_gateT, 256, 512);
  packT(w_out, w_outT, 512, 256);

  frames_kernel<<<(N + 255) / 256, 256, 0, stream>>>(pos, Rg, cag, N);
  qkv_kernel<<<N / 64, 256, 0, stream>>>(x, wqT, wkT, wvT, q, kb, vb, N);
  pair_attn_kernel<<<N / 2, 128, 0, stream>>>(
      pos, maskp, nbrs, resi, chain, batch, Rg, cag, w_relpos, w_dist, w_dir, w_rot,
      w_pvec, pln_s, pln_b, w_p1T, b_p1, w_p2T, b_p2, w_biasT, w_pvT, q, kb, vb, ao, N);
  outproj_kernel<<<N / 32, 256, 0, stream>>>(ao, woT, bo, x, finc, maskp, ln1_s, ln1_b,
                                             out2, x1ln, N);
  transition_kernel<<<N / 32, 256, 0, stream>>>(x1ln, pos, Rg, cag, wu1, bu1, wu2T, bu2,
                                                w_updT, w_gateT, w_outT, b_out, out2, ao, N);
  ln2_kernel<<<N / 8, 256, 0, stream>>>(ao, ln2_s, ln2_b, out1, N);
}
